// SLSA_87840671137898
// MI455X (gfx1250) — matrix operand order repaired and checked by reading, not compiled, not run on hardware
//
#include <hip/hip_runtime.h>
#include <hip/hip_bf16.h>

#define BB 4
#define TT 2048
#define CC 1024
#define HH 16
#define HS 64

typedef _Float16 v8h  __attribute__((ext_vector_type(8)));
typedef _Float16 v16h __attribute__((ext_vector_type(16)));
typedef float    v8f  __attribute__((ext_vector_type(8)));

static __device__ __forceinline__ v8f wmma_f16(v16h a, v16h b, v8f c) {
  // D = A(16x32 f16) * B(32x16 f16) + C(16x16 f32)
  return __builtin_amdgcn_wmma_f32_16x16x32_f16(false, a, false, b, (short)0, c,
                                                false, false);
}

// One async DMA of 16B global -> 16B LDS (ASYNCcnt-tracked).
static __device__ __forceinline__ void async_b128(unsigned lds_addr,
                                                  const void* gaddr) {
  asm volatile("global_load_async_to_lds_b128 %0, %1, off"
               :: "v"(lds_addr), "v"(gaddr) : "memory");
}
// 32B (two B128, shared address setup; offset applies to both LDS and global).
static __device__ __forceinline__ void async_b256(unsigned lds_addr,
                                                  const void* gaddr) {
  asm volatile("global_load_async_to_lds_b128 %0, %1, off\n\t"
               "global_load_async_to_lds_b128 %0, %1, off offset:16"
               :: "v"(lds_addr), "v"(gaddr) : "memory");
}
static __device__ __forceinline__ void wait_async_le1() {
  asm volatile("s_wait_asynccnt 0x1" ::: "memory");
}
static __device__ __forceinline__ void wait_async_le4() {
  asm volatile("s_wait_asynccnt 0x4" ::: "memory");
}
static __device__ __forceinline__ void wait_async_0() {
  asm volatile("s_wait_asynccnt 0x0" ::: "memory");
}
static __device__ __forceinline__ unsigned lds_off(const void* p) {
  return (unsigned)(unsigned long long)p;  // low 32 bits = LDS byte offset
}

// A-fragment: lane holds row (lane%16); halves 0..15 -> K = 16*(lane/16)+i.
static __device__ __forceinline__ v16h load_afrag(const _Float16* row, int g) {
  v8h lo = *(const v8h*)(row + 16 * g);
  v8h hi = *(const v8h*)(row + 16 * g + 8);
  v16h r;
#pragma unroll
  for (int i = 0; i < 8; ++i) { r[i] = lo[i]; r[i + 8] = hi[i]; }
  return r;
}

__global__ void cvt_f32_to_f16(const float* __restrict__ in,
                               _Float16* __restrict__ out, int n) {
  int i = blockIdx.x * blockDim.x + threadIdx.x;
  if (i < n) out[i] = (_Float16)in[i];
}

// Wt[n*K+k] = W[k*N+n] (+ identity if addI) as f16
__global__ void transpose_cvt(const float* __restrict__ W,
                              _Float16* __restrict__ Wt, int K, int N, int addI) {
  int i = blockIdx.x * blockDim.x + threadIdx.x;
  if (i >= K * N) return;
  int n = i / K, k = i % K;
  float v = W[(size_t)k * N + n];
  if (addI && k == n) v += 1.0f;
  Wt[(size_t)n * K + k] = (_Float16)v;
}

// out[M,N] = A[M,K](f16) @ Wt[N,K]^T(f16) + bias.
// Block = 8 waves = 128x64 macro-tile. Weight slab (64xN x 32xK = 4KB) staged
// into double-buffered LDS by async DMA; A fragments software-pipelined.
template <bool F16OUT>
__global__ void gemm_wmma(const _Float16* __restrict__ A,
                          const _Float16* __restrict__ Wt,
                          const float* __restrict__ bias, void* __restrict__ out,
                          int M, int N, int K) {
  alignas(64) __shared__ _Float16 lds_w[2][64 * 32];
  int tid  = threadIdx.x;
  int wv   = tid >> 5;
  int lane = tid & 31;
  int g = lane >> 4, c = lane & 15;
  int n0 = blockIdx.x * 64;
  int m0 = blockIdx.y * 128 + wv * 16;

  // per-thread DMA mapping: row n = tid/4 (0..63), k-chunk = (tid%4)*8 halves
  int wn = tid >> 2, wc = (tid & 3) * 8;
  const _Float16* gsrc = Wt + (size_t)(n0 + wn) * K + wc;
  unsigned ldst0 = lds_off(&lds_w[0][wn * 32 + wc]);
  unsigned ldst1 = lds_off(&lds_w[1][wn * 32 + wc]);

  async_b128(ldst0, gsrc);          // slab k0 = 0
  async_b128(ldst1, gsrc + 32);     // slab k0 = 32

  const _Float16* arow = A + (size_t)(m0 + c) * K;
  v8f acc[4] = {};
  int nk = K >> 5;
  v16h a = load_afrag(arow, g);     // pipelined A fragment
  for (int i = 0; i < nk; ++i) {
    int k0 = i << 5;
    wait_async_le1();      // current slab landed (in-order completion)
    __syncthreads();       // whole slab visible to all waves
    v16h acur = a;
    if (i + 1 < nk) a = load_afrag(arow + k0 + 32, g);  // prefetch next
    const _Float16* wbuf = lds_w[i & 1];
#pragma unroll
    for (int j = 0; j < 4; ++j) {
      v16h b = *(const v16h*)(wbuf + (j * 16 + c) * 32 + 16 * g);
      acc[j] = wmma_f16(acur, b, acc[j]);
    }
    __syncthreads();       // all waves done reading before overwrite
    if (k0 + 64 < K) async_b128(i & 1 ? ldst1 : ldst0, gsrc + k0 + 64);
  }
#pragma unroll
  for (int j = 0; j < 4; ++j) {
    int col = n0 + j * 16 + c;
    float bv = bias ? bias[col] : 0.0f;
#pragma unroll
    for (int r = 0; r < 8; ++r) {
      int row = m0 + r + 8 * g;
      float v = acc[j][r] + bv;
      if (F16OUT) ((_Float16*)out)[(size_t)row * N + col] = (_Float16)v;
      else        ((float*)out)[(size_t)row * N + col] = v;
    }
  }
}

// v = (v0 @ (I+W1) + b1) * sigmoid(v0 @ W2 + b2), per head; output transposed
// vT[B,H,HS,T]. (I+W1)^T and W2^T (8KB each) are async-DMA'd into LDS once
// per block and shared by all 8 waves.
__global__ void gated_v_kernel(const _Float16* __restrict__ v0,
                               const _Float16* __restrict__ W1t,
                               const _Float16* __restrict__ W2t,
                               const float* __restrict__ b1,
                               const float* __restrict__ b2,
                               _Float16* __restrict__ vT) {
  alignas(64) __shared__ _Float16 lds_w12[2 * 64 * 64];  // W1t | W2t
  int tid  = threadIdx.x;
  int lane = tid & 31;
  int g = lane >> 4, c = lane & 15;

  // one 64B chunk per thread: matrix = tid/128, row = (tid/2)%64, half = tid%2
  {
    int mtx = tid >> 7, row = (tid >> 1) & 63, half = tid & 1;
    const _Float16* src = (mtx ? W2t : W1t) + row * 64 + half * 32;
    unsigned dst = lds_off(&lds_w12[mtx * 4096 + row * 64 + half * 32]);
    async_b256(dst, src);
    async_b256(dst + 32, src + 16);
  }

  int wid  = blockIdx.x * (blockDim.x >> 5) + (tid >> 5);
  int hdr = wid / (BB * TT / 16);
  int rt  = wid % (BB * TT / 16);
  int bt0 = rt * 16;
  const _Float16* arow = v0 + (size_t)(bt0 + c) * CC + hdr * HS;
  v16h a0 = load_afrag(arow, g);
  v16h a1f = load_afrag(arow + 32, g);

  wait_async_0();
  __syncthreads();
  const _Float16* w1 = lds_w12;
  const _Float16* w2 = lds_w12 + 4096;

  v8f a1[4] = {}, a2[4] = {};
#pragma unroll
  for (int k0 = 0; k0 < HS; k0 += 32) {
    v16h a = (k0 == 0) ? a0 : a1f;
#pragma unroll
    for (int j = 0; j < 4; ++j) {
      v16h bw1 = *(const v16h*)(w1 + (j * 16 + c) * 64 + k0 + 16 * g);
      v16h bw2 = *(const v16h*)(w2 + (j * 16 + c) * 64 + k0 + 16 * g);
      a1[j] = wmma_f16(a, bw1, a1[j]);
      a2[j] = wmma_f16(a, bw2, a2[j]);
    }
  }
#pragma unroll
  for (int j = 0; j < 4; ++j) {
    int d = j * 16 + c;
    float bb1 = b1[d], bb2 = b2[d];
#pragma unroll
    for (int r = 0; r < 8; ++r) {
      int bt = bt0 + r + 8 * g;
      int b = bt >> 11, t = bt & (TT - 1);
      float av = a1[j][r] + bb1;
      float gv = a2[j][r] + bb2;
      float o  = av * (1.0f / (1.0f + __expf(-gv)));
      vT[((size_t)(b * HH + hdr) * HS + d) * TT + t] = (_Float16)o;
    }
  }
}

// Flash attention. Block = 4 waves = 4 adjacent 16-query tiles of one (b,h).
// K-slab (32 keys x 64 d) and V-slab (64 d x 32 keys) are async-DMA'd into
// double-buffered LDS shared by all 4 waves; block-uniform k-loop (fully
// masked tiles are exact no-ops for the online softmax: alpha=1, p=0).
__global__ void flash_attn_kernel(const _Float16* __restrict__ uh,
                                  const _Float16* __restrict__ xh,
                                  const _Float16* __restrict__ vT,
                                  _Float16* __restrict__ yh) {
  alignas(64) __shared__ _Float16 lds_k[2][32 * 64];
  alignas(64) __shared__ _Float16 lds_v[2][64 * 32];
  alignas(64) __shared__ _Float16 lds_p[4][16 * 40];
  int tid  = threadIdx.x;
  int wv   = tid >> 5;
  int lane = tid & 31;
  int g = lane >> 4, c = lane & 15;
  int b = blockIdx.x / HH, h = blockIdx.x % HH;
  int q0   = (blockIdx.y * 4 + wv) * 16;
  int kmax = (blockIdx.y * 4 + 3) * 16 + 16;   // block-uniform bound (>= 64)
  const float scale = 0.03125f;                // 1/sqrt(C) = 1/32

  // DMA mapping: K-slab: key = tid/4, 32B chunk = tid%4 (2 B128 each)
  //              V-slab: d = tid/2,  32B chunk = tid%2 (2 B128 each)
  int kkey = tid >> 2, kch = (tid & 3) * 16;
  int vd   = tid >> 1, vch = (tid & 1) * 16;
  const _Float16* gk = xh + (size_t)(b * TT + kkey) * CC + h * HS + kch;
  const _Float16* gv = vT + ((size_t)(b * HH + h) * HS + vd) * TT + vch;
  unsigned lk[2] = { lds_off(&lds_k[0][kkey * 64 + kch]),
                     lds_off(&lds_k[1][kkey * 64 + kch]) };
  unsigned lv[2] = { lds_off(&lds_v[0][vd * 32 + vch]),
                     lds_off(&lds_v[1][vd * 32 + vch]) };

  // prologue: two slabs in flight (kmax >= 64 always)
  async_b256(lk[0], gk);                         // keys 0..31
  async_b256(lv[0], gv);
  async_b256(lk[1], gk + (size_t)32 * CC);       // keys 32..63
  async_b256(lv[1], gv + 32);

  const _Float16* urow = uh + (size_t)(b * TT + q0 + c) * CC + h * HS;
  v16h u0 = load_afrag(urow, g);
  v16h u1 = load_afrag(urow + 32, g);

  v8f accO[4] = {};
  float m[8], ell[8];
#pragma unroll
  for (int r = 0; r < 8; ++r) { m[r] = -1e30f; ell[r] = 0.0f; }

  int nk = kmax >> 5;
  for (int i = 0; i < nk; ++i) {
    int k0 = i << 5;
    wait_async_le4();   // this slab's 4 per-thread DMAs done (in-order)
    __syncthreads();
    const _Float16* kbuf = lds_k[i & 1];
    const _Float16* vbuf = lds_v[i & 1];

    v8f s[2];
#pragma unroll
    for (int t = 0; t < 2; ++t) {
      v16h bk0 = *(const v16h*)(kbuf + (t * 16 + c) * 64 + 16 * g);
      v16h bk1 = *(const v16h*)(kbuf + (t * 16 + c) * 64 + 32 + 16 * g);
      v8f z = {};
      z = wmma_f16(u0, bk0, z);
      z = wmma_f16(u1, bk1, z);
      s[t] = z;
    }
    // scale + causal mask
#pragma unroll
    for (int t = 0; t < 2; ++t) {
      int key = k0 + t * 16 + c;
#pragma unroll
      for (int r = 0; r < 8; ++r) {
        int q = q0 + r + 8 * g;
        float sv = s[t][r] * scale;
        s[t][r] = (key <= q) ? sv : -1e30f;
      }
    }
    // online softmax (row r lives in one 16-lane group)
#pragma unroll
    for (int r = 0; r < 8; ++r) {
      float mr = fmaxf(s[0][r], s[1][r]);
#pragma unroll
      for (int off = 8; off >= 1; off >>= 1)
        mr = fmaxf(mr, __shfl_xor(mr, off, 32));
      float mn = fmaxf(m[r], mr);
      float alpha = __expf(m[r] - mn);
      float p0 = __expf(s[0][r] - mn);
      float p1 = __expf(s[1][r] - mn);
      float rs = p0 + p1;
#pragma unroll
      for (int off = 8; off >= 1; off >>= 1)
        rs += __shfl_xor(rs, off, 32);
      ell[r] = ell[r] * alpha + rs;
      m[r] = mn;
#pragma unroll
      for (int j = 0; j < 4; ++j) accO[j][r] *= alpha;
      int row = r + 8 * g;
      lds_p[wv][row * 40 + c]      = (_Float16)p0;
      lds_p[wv][row * 40 + 16 + c] = (_Float16)p1;
    }
    __builtin_amdgcn_wave_barrier();  // same-wave LDS RAW: DS is in-order
    const _Float16* pr = &lds_p[wv][c * 40];
    v8h plo = *(const v8h*)(pr + 16 * g);
    v8h phi = *(const v8h*)(pr + 16 * g + 8);
    v16h pa;
#pragma unroll
    for (int i2 = 0; i2 < 8; ++i2) { pa[i2] = plo[i2]; pa[i2 + 8] = phi[i2]; }
    __builtin_amdgcn_wave_barrier();
#pragma unroll
    for (int j = 0; j < 4; ++j) {
      v16h bvf = *(const v16h*)(vbuf + (j * 16 + c) * 32 + 16 * g);
      accO[j] = wmma_f16(pa, bvf, accO[j]);
    }
    __syncthreads();   // all waves done with this slab before refill
    if (k0 + 64 < kmax) {
      int bi = i & 1;
      async_b256(lk[bi], gk + (size_t)(k0 + 64) * CC);
      async_b256(lv[bi], gv + k0 + 64);
    }
  }
  // normalize and store y (f16, [B,T,C] layout)
#pragma unroll
  for (int j = 0; j < 4; ++j) {
#pragma unroll
    for (int r = 0; r < 8; ++r) {
      float o = accO[j][r] / ell[r];
      yh[(size_t)(b * TT + q0 + r + 8 * g) * CC + h * HS + j * 16 + c] = (_Float16)o;
    }
  }
}

extern "C" void kernel_launch(void* const* d_in, const int* in_sizes, int n_in,
                              void* d_out, int out_size, void* d_ws, size_t ws_size,
                              hipStream_t stream) {
  const float* x  = (const float*)d_in[0];
  const float* Wu = (const float*)d_in[1];
  const float* bu = (const float*)d_in[2];
  const float* Wv = (const float*)d_in[3];
  const float* bv = (const float*)d_in[4];
  const float* W1 = (const float*)d_in[5];
  const float* b1 = (const float*)d_in[6];
  const float* W2 = (const float*)d_in[7];
  const float* b2 = (const float*)d_in[8];
  const float* Wp = (const float*)d_in[9];
  const float* bp = (const float*)d_in[10];
  float* out = (float*)d_out;

  char* ws = (char*)d_ws;
  size_t off = 0;
  auto alloc = [&](size_t bytes) {
    void* p = ws + off;
    off += (bytes + 255) & ~(size_t)255;
    return p;
  };
  const int NTC = BB * TT * CC;  // 8,388,608
  _Float16* xh  = (_Float16*)alloc((size_t)NTC * 2);
  _Float16* uh  = (_Float16*)alloc((size_t)NTC * 2);
  _Float16* v0h = (_Float16*)alloc((size_t)NTC * 2);
  _Float16* vTh = (_Float16*)alloc((size_t)NTC * 2);
  _Float16* yh  = (_Float16*)alloc((size_t)NTC * 2);
  _Float16* Wut = (_Float16*)alloc((size_t)CC * CC * 2);
  _Float16* Wvt = (_Float16*)alloc((size_t)CC * CC * 2);
  _Float16* Wpt = (_Float16*)alloc((size_t)CC * CC * 2);
  _Float16* W1t = (_Float16*)alloc((size_t)HS * HS * 2);
  _Float16* W2t = (_Float16*)alloc((size_t)HS * HS * 2);

  cvt_f32_to_f16<<<(NTC + 255) / 256, 256, 0, stream>>>(x, xh, NTC);
  transpose_cvt<<<(CC * CC + 255) / 256, 256, 0, stream>>>(Wu, Wut, CC, CC, 0);
  transpose_cvt<<<(CC * CC + 255) / 256, 256, 0, stream>>>(Wv, Wvt, CC, CC, 0);
  transpose_cvt<<<(CC * CC + 255) / 256, 256, 0, stream>>>(Wp, Wpt, CC, CC, 0);
  transpose_cvt<<<(HS * HS + 255) / 256, 256, 0, stream>>>(W1, W1t, HS, HS, 1);
  transpose_cvt<<<(HS * HS + 255) / 256, 256, 0, stream>>>(W2, W2t, HS, HS, 0);

  const int M = BB * TT;  // 8192
  dim3 ggrid(CC / 64, M / 128);  // (16, 64)
  gemm_wmma<true><<<ggrid, 256, 0, stream>>>(xh, Wut, bu, uh, M, CC, CC);
  gemm_wmma<true><<<ggrid, 256, 0, stream>>>(xh, Wvt, bv, v0h, M, CC, CC);

  gated_v_kernel<<<(HH * (M / 16)) / 8, 256, 0, stream>>>(v0h, W1t, W2t, b1, b2, vTh);

  flash_attn_kernel<<<dim3(BB * HH, TT / 64), 128, 0, stream>>>(uh, xh, vTh, yh);

  gemm_wmma<false><<<ggrid, 256, 0, stream>>>(yh, Wpt, bp, (void*)out, M, CC, CC);
}
